// TopKSparseMoE_9431748182291
// MI455X (gfx1250) — matrix operand order repaired and checked by reading, not compiled, not run on hardware
//
#include <hip/hip_runtime.h>
#include <hip/hip_bf16.h>
#include <stdint.h>

// Problem constants (from reference)
#define B_  8192
#define D_  1024
#define H_  4096
#define O_  1024
#define E_  8
#define K_  2
#define TILE_ 16
#define MAXSLOTS_ (B_*K_ + E_*TILE_)        // 16512 (per-expert padded to 16)
#define MAXTILES_ (B_*K_/TILE_ + E_)        // 1032

typedef __attribute__((ext_vector_type(16))) __bf16 v16bf;
typedef __attribute__((ext_vector_type(8)))  float  v8f;
typedef unsigned int v4u __attribute__((ext_vector_type(4)));
typedef int v8i __attribute__((ext_vector_type(8)));
typedef int v4i __attribute__((ext_vector_type(4)));

union Frag { v16bf v; uint4 q[2]; };

// f32 -> bf16, round-to-nearest-even
__device__ __forceinline__ unsigned short f2bf(float f) {
  unsigned u = __float_as_uint(f);
  u += 0x7fffu + ((u >> 16) & 1u);
  return (unsigned short)(u >> 16);
}

// ---- Tensor Data Mover: async DMA of a [32 rows x 128 cols] f32 tile ----
// D# per CDNA5 ISA ch.8: group0 {count=1|load|user, lds_addr, global_addr, type=2}
// group1 {data_size=4B, tensor_dim0=128, tensor_dim1=32, tile_dim0=128,
//         tile_dim1=32, tensor_dim0_stride=stride_elems}
// This toolchain exposes the 6-arg builtin: (v4u, v8i, v4i, v4i, v8i, i32 cpol)
__device__ __forceinline__ void tdm_load_2d(const void* lds_dst, const void* gsrc,
                                            unsigned stride_elems) {
  unsigned lds_off = (unsigned)(unsigned long long)(uintptr_t)lds_dst; // addr[31:0] = LDS offset
  unsigned long long ga = (unsigned long long)(uintptr_t)gsrc;
  v4u g0 = { 1u,                                   // count=1, is_restore=0, load
             lds_off,                              // lds_addr
             (unsigned)ga,                         // global_addr[31:0]
             (unsigned)((ga >> 32) & 0x01ffffffu) | (2u << 30) }; // addr[56:32] | type=2
  v8i g1 = { (int)(2u << 16),                      // data_size=4B (code 2)
             (int)(128u << 16),                    // tensor_dim0 = 128 (bits 63:48)
             (int)(32u << 16),                     // tensor_dim1 = 32  (bits 95:80)
             (int)(128u << 16),                    // tile_dim0 = 128   (bits 127:112)
             32,                                   // tile_dim1=32, tile_dim2=0
             (int)stride_elems,                    // tensor_dim0_stride lo32
             0, 0 };
  v4i gz4 = { 0, 0, 0, 0 };
  v8i gz8 = { 0, 0, 0, 0, 0, 0, 0, 0 };
  __builtin_amdgcn_tensor_load_to_lds(g0, g1, gz4, gz4, gz8, 0);
}

// ---------------- zero output / small ws counters ----------------
__global__ void k_zero(float* __restrict__ outp, float* __restrict__ imp,
                       int* __restrict__ counts, size_t n) {
  size_t gid = (size_t)blockIdx.x * blockDim.x + threadIdx.x;
  if (gid < n) outp[gid] = 0.f;
  if (gid < E_) { imp[gid] = 0.f; counts[gid] = 0; }
}

// ---------------- gating: logits, top-2, softmax, stats ----------------
__global__ __launch_bounds__(256)
void k_gate(const float* __restrict__ x, const float* __restrict__ Wg,
            const float* __restrict__ bg,
            float* __restrict__ gates, float* __restrict__ topi_o,
            float* __restrict__ imp, int* __restrict__ counts,
            int* __restrict__ topE, float* __restrict__ topW) {
  __shared__ float lg[32][E_ + 1];
  const int tid = threadIdx.x;
  const int rl = tid >> 3, e = tid & 7;        // 32 rows x 8 experts per block
  const int row = blockIdx.x * 32 + rl;
  float acc = bg[e];
  const float* xr = x + (size_t)row * D_;
  for (int d = 0; d < D_; ++d) acc += xr[d] * Wg[d * E_ + e];
  lg[rl][e] = acc;
  __syncthreads();
  if (tid < 32) {
    const int r = blockIdx.x * 32 + tid;
    float v[E_];
    #pragma unroll
    for (int j = 0; j < E_; ++j) v[j] = lg[tid][j];
    int i1 = 0;
    #pragma unroll
    for (int j = 1; j < E_; ++j) if (v[j] > v[i1]) i1 = j;   // lowest index wins ties
    int i2 = -1;
    #pragma unroll
    for (int j = 0; j < E_; ++j)
      if (j != i1 && (i2 < 0 || v[j] > v[i2])) i2 = j;
    float e2 = expf(v[i2] - v[i1]);
    float s = 1.f + e2;
    float g1 = 1.f / s, g2 = e2 / s;
    #pragma unroll
    for (int j = 0; j < E_; ++j)
      gates[(size_t)r * E_ + j] = (j == i1) ? g1 : ((j == i2) ? g2 : 0.f);
    topi_o[(size_t)r * K_ + 0] = (float)i1;
    topi_o[(size_t)r * K_ + 1] = (float)i2;
    atomicAdd(&imp[i1], g1); atomicAdd(&imp[i2], g2);
    atomicAdd(&counts[i1], 1); atomicAdd(&counts[i2], 1);
    topE[r * K_ + 0] = i1; topE[r * K_ + 1] = i2;
    topW[r * K_ + 0] = g1; topW[r * K_ + 1] = g2;
  }
}

// ---------------- tile offsets from per-expert counts ----------------
__global__ void k_route(const int* __restrict__ counts, int* __restrict__ cursor,
                        int* __restrict__ tileExpert, int* __restrict__ tileStart,
                        int* __restrict__ assignRow) {
  __shared__ int tileBase[E_ + 1], slotBase[E_ + 1];
  if (threadIdx.x == 0) {
    int tb = 0, sb = 0;
    for (int e = 0; e < E_; ++e) {
      tileBase[e] = tb; slotBase[e] = sb;
      int pt = (counts[e] + TILE_ - 1) / TILE_;
      tb += pt; sb += pt * TILE_;
      cursor[e] = slotBase[e];
    }
    tileBase[E_] = tb; slotBase[E_] = sb;
  }
  __syncthreads();
  for (int t = threadIdx.x; t < MAXTILES_; t += blockDim.x) {
    int e = -1;
    #pragma unroll
    for (int j = 0; j < E_; ++j)
      if (t >= tileBase[j] && t < tileBase[j + 1]) e = j;
    tileExpert[t] = e;
    tileStart[t] = (e >= 0) ? slotBase[e] + (t - tileBase[e]) * TILE_ : 0;
  }
  for (int s = threadIdx.x; s < MAXSLOTS_; s += blockDim.x) assignRow[s] = -1;
}

// ---------------- scatter rows into expert-sorted slot list ----------------
__global__ void k_scatter(const int* __restrict__ topE, const float* __restrict__ topW,
                          int* __restrict__ cursor,
                          int* __restrict__ assignRow, float* __restrict__ assignW) {
  int row = blockIdx.x * blockDim.x + threadIdx.x;
  if (row >= B_) return;
  #pragma unroll
  for (int k = 0; k < K_; ++k) {
    int e = topE[row * K_ + k];
    int pos = atomicAdd(&cursor[e], 1);
    assignRow[pos] = row;
    assignW[pos] = topW[row * K_ + k];
  }
}

// ---------------- fused 2-layer grouped GEMM per 16-token tile ----------------
// 8 waves; weights streamed by the TDM (double-buffered f32 LDS staging) and
// converted in LDS to bf16 [n][k] for WMMA. Fragments per CDNA5 ISA layouts:
//   A (16-bit 16x32): lanes0-15 row M=l, K=kh*8+{0..7} in q[0], 16+kh*8+{0..7} in q[1]
//   B (32x16):        lane col N=l&15, K=kh*8+{0..7} in q[0], 16+kh*8+{0..7} in q[1]
//   C (f32 16x16):    VGPR r -> M=r+kh*8, N=l&15
__global__ __launch_bounds__(256)
void moe_ffn_kernel(const float* __restrict__ x,
                    const float* __restrict__ W1, const float* __restrict__ b1,
                    const float* __restrict__ W2, const float* __restrict__ b2,
                    const int* __restrict__ tileExpert, const int* __restrict__ tileStart,
                    const int* __restrict__ assignRow, const float* __restrict__ assignW,
                    float* __restrict__ outp) {
  __shared__ __align__(16) unsigned short xs[TILE_][D_ + 8];   // 33 KB, K-contig rows
  __shared__ __align__(16) unsigned short hsb[TILE_][128 + 8]; // 4.3 KB hidden chunk
  __shared__ __align__(16) unsigned short wst[128][40];        // 10 KB bf16 W [n][k]
  __shared__ __align__(16) float pingf[2][32 * 128];           // 32 KB TDM staging
  __shared__ int   rowIdx[TILE_];
  __shared__ float rowW[TILE_];

  const int e = tileExpert[blockIdx.x];
  if (e < 0) return;                           // uniform per block
  const int s0 = tileStart[blockIdx.x];
  const int tid = threadIdx.x;

  if (tid < TILE_) {
    int r = assignRow[s0 + tid];
    rowIdx[tid] = (r < 0) ? 0 : r;
    rowW[tid]   = (r < 0) ? 0.f : assignW[s0 + tid];
  }
  __syncthreads();

  // stage x tile -> bf16 LDS
  for (int idx = tid; idx < TILE_ * D_; idx += 256) {
    int r = idx / D_, d = idx % D_;
    xs[r][d] = f2bf(x[(size_t)rowIdx[r] * D_ + d]);
  }

  const float* W1e = W1 + (size_t)e * D_ * H_;
  const float* W2e = W2 + (size_t)e * H_ * O_;
  const float* b1e = b1 + (size_t)e * H_;
  const float* b2e = b2 + (size_t)e * O_;

  const int wave = tid >> 5;
  const int lane = tid & 31;
  const int nl = lane & 15;
  const int kh = lane >> 4;
  const int bn = wave * 16 + nl;
  const bool issuer = (wave == 0);             // one wave drives the TDM

  const v8f vzero = {0.f,0.f,0.f,0.f,0.f,0.f,0.f,0.f};
  v8f acc2[8];
  #pragma unroll
  for (int i = 0; i < 8; ++i) acc2[i] = vzero;

  __syncthreads();

  for (int hb = 0; hb < H_; hb += 128) {
    // ---- layer 1: acc1 = x_tile @ W1[:, hb:hb+128] ----
    v8f acc1 = vzero;
    if (issuer) tdm_load_2d(&pingf[0][0], W1e + hb, H_);
    for (int i = 0; i < 32; ++i) {             // 32 K-chunks over D
      const int buf = i & 1;
      if (issuer) {
        if (i + 1 < 32) {
          tdm_load_2d(&pingf[buf ^ 1][0], W1e + (size_t)(i + 1) * 32 * H_ + hb, H_);
          __builtin_amdgcn_s_wait_tensorcnt(1);  // chunk i landed (in-order TDM)
        } else {
          __builtin_amdgcn_s_wait_tensorcnt(0);
        }
      }
      __syncthreads();
      // convert f32 chunk [32][128] -> bf16 transposed wst[n][k], packed stores
      for (int idx = tid; idx < 2048; idx += 256) {
        int n = idx & 127, kp = idx >> 7;
        float f0 = pingf[buf][(2 * kp)     * 128 + n];
        float f1 = pingf[buf][(2 * kp + 1) * 128 + n];
        unsigned pk = (unsigned)f2bf(f0) | ((unsigned)f2bf(f1) << 16);
        *(unsigned*)&wst[n][2 * kp] = pk;
      }
      __syncthreads();
      Frag a, b;
      const int kb = i * 32;
      a.q[0] = *(const uint4*)&xs[nl][kb + kh * 8];
      a.q[1] = *(const uint4*)&xs[nl][kb + kh * 8 + 16];
      b.q[0] = *(const uint4*)&wst[bn][kh * 8];
      b.q[1] = *(const uint4*)&wst[bn][kh * 8 + 16];
      acc1 = __builtin_amdgcn_wmma_f32_16x16x32_bf16(false, a.v, false, b.v,
                                                     (short)0, acc1, false, false);
    }
    __syncthreads();
    // bias + ReLU -> hidden chunk in LDS (bf16)
    {
      const int col = wave * 16 + nl;
      float bias = b1e[hb + col];
      #pragma unroll
      for (int r8 = 0; r8 < 8; ++r8) {
        float hv = acc1[r8] + bias;
        hsb[r8 + kh * 8][col] = f2bf(hv > 0.f ? hv : 0.f);
      }
    }
    __syncthreads();
    // ---- layer 2 partial: out += h_chunk @ W2[hb:hb+128, :] ----
    // 32 flattened chunks: j -> (oi = j>>2 output 128-block, kb2 = (j&3)*32)
    if (issuer) tdm_load_2d(&pingf[0][0], W2e + (size_t)hb * O_, O_);
    #pragma unroll
    for (int j = 0; j < 32; ++j) {
      const int buf = j & 1;
      const int oi = j >> 2, kb2 = (j & 3) * 32;
      if (issuer) {
        if (j + 1 < 32) {
          const int oj = j + 1;
          tdm_load_2d(&pingf[buf ^ 1][0],
                      W2e + (size_t)(hb + (oj & 3) * 32) * O_ + (oj >> 2) * 128, O_);
          __builtin_amdgcn_s_wait_tensorcnt(1);
        } else {
          __builtin_amdgcn_s_wait_tensorcnt(0);
        }
      }
      __syncthreads();
      for (int idx = tid; idx < 2048; idx += 256) {
        int n = idx & 127, kp = idx >> 7;
        float f0 = pingf[buf][(2 * kp)     * 128 + n];
        float f1 = pingf[buf][(2 * kp + 1) * 128 + n];
        unsigned pk = (unsigned)f2bf(f0) | ((unsigned)f2bf(f1) << 16);
        *(unsigned*)&wst[n][2 * kp] = pk;
      }
      __syncthreads();
      Frag a, b;
      a.q[0] = *(const uint4*)&hsb[nl][kb2 + kh * 8];
      a.q[1] = *(const uint4*)&hsb[nl][kb2 + kh * 8 + 16];
      b.q[0] = *(const uint4*)&wst[bn][kh * 8];
      b.q[1] = *(const uint4*)&wst[bn][kh * 8 + 16];
      acc2[oi] = __builtin_amdgcn_wmma_f32_16x16x32_bf16(false, a.v, false, b.v,
                                                         (short)0, acc2[oi],
                                                         false, false);
    }
    __syncthreads();   // wst/hsb safe to reuse next hb
  }

  // ---- gate-weighted combine: out[row] += w * (expert_out + b2) ----
  #pragma unroll
  for (int oi = 0; oi < 8; ++oi) {
    const int col = oi * 128 + wave * 16 + nl;
    float bias = b2e[col];
    #pragma unroll
    for (int r8 = 0; r8 < 8; ++r8) {
      const int row = r8 + kh * 8;
      float w = rowW[row];
      if (w != 0.f)
        atomicAdd(&outp[(size_t)rowIdx[row] * O_ + col], w * (acc2[oi][r8] + bias));
    }
  }
}

// ---------------- finalize load / importance ----------------
__global__ void k_finalize(const float* __restrict__ imp,
                           float* __restrict__ load_o, float* __restrict__ imp_o) {
  int e = threadIdx.x;
  if (e < E_) {
    load_o[e] = imp[e] / (float)B_;
    imp_o[e]  = imp[e];
  }
}

extern "C" void kernel_launch(void* const* d_in, const int* in_sizes, int n_in,
                              void* d_out, int out_size, void* d_ws, size_t ws_size,
                              hipStream_t stream) {
  (void)in_sizes; (void)n_in; (void)out_size; (void)ws_size;
  const float* x  = (const float*)d_in[0];
  const float* Wg = (const float*)d_in[1];
  const float* bg = (const float*)d_in[2];
  const float* W1 = (const float*)d_in[3];
  const float* b1 = (const float*)d_in[4];
  const float* W2 = (const float*)d_in[5];
  const float* b2 = (const float*)d_in[6];

  // d_out: output[B*O] | gates[B*E] | load[E] | importance[E] | top_i[B*K] (floats)
  float* out_f  = (float*)d_out;
  float* gates  = out_f + (size_t)B_ * O_;
  float* load_o = gates + (size_t)B_ * E_;
  float* imp_o  = load_o + E_;
  float* topi_o = imp_o + E_;

  // workspace layout (~273 KB)
  char* ws = (char*)d_ws;
  float* imp        = (float*)(ws + 0);
  int*   counts     = (int*)(ws + 32);
  int*   cursor     = (int*)(ws + 64);
  int*   tileExpert = (int*)(ws + 96);
  int*   tileStart  = tileExpert + MAXTILES_;
  int*   topE       = (int*)(ws + 16384);
  float* topW       = (float*)(ws + 16384 + 4 * B_ * K_);
  int*   assignRow  = (int*)(ws + 16384 + 8 * B_ * K_);
  float* assignW    = (float*)(ws + 16384 + 8 * B_ * K_ + 4 * MAXSLOTS_);

  k_zero<<<((size_t)B_ * O_ + 255) / 256, 256, 0, stream>>>(out_f, imp, counts,
                                                            (size_t)B_ * O_);
  k_gate<<<B_ / 32, 256, 0, stream>>>(x, Wg, bg, gates, topi_o, imp, counts,
                                      topE, topW);
  k_route<<<1, 256, 0, stream>>>(counts, cursor, tileExpert, tileStart, assignRow);
  k_scatter<<<B_ / 256, 256, 0, stream>>>(topE, topW, cursor, assignRow, assignW);
  moe_ffn_kernel<<<MAXTILES_, 256, 0, stream>>>(x, W1, b1, W2, b2, tileExpert,
                                                tileStart, assignRow, assignW, out_f);
  k_finalize<<<1, 32, 0, stream>>>(imp, load_o, imp_o);
}
